// GCN_Body_56006373539865
// MI455X (gfx1250) — matrix operand order repaired and checked by reading, not compiled, not run on hardware
//
#include <hip/hip_runtime.h>

typedef __attribute__((ext_vector_type(16))) _Float16 v16h;
typedef __attribute__((ext_vector_type(8)))  _Float16 v8h;
typedef __attribute__((ext_vector_type(8)))  float    v8f;
typedef __attribute__((ext_vector_type(4)))  float    v4f;

#define NFEAT 128
#define LDS_PITCH 136   // 128 + 8 halfs pad to de-align LDS banks

// ---------------- small utility kernels ----------------

__global__ void zero_f32_kernel(float* __restrict__ p, int n) {
  int i = blockIdx.x * blockDim.x + threadIdx.x;
  if (i < n) p[i] = 0.0f;
}

__global__ void init_deg_kernel(float* __restrict__ deg, int n) {
  int i = blockIdx.x * blockDim.x + threadIdx.x;
  if (i < n) deg[i] = 1.0f;  // self-loop contributes 1
}

__global__ void count_deg_kernel(const long long* __restrict__ col,
                                 float* __restrict__ deg, int e) {
  int i = blockIdx.x * blockDim.x + threadIdx.x;
  if (i < e) atomicAdd(&deg[(int)col[i]], 1.0f);
}

__global__ void rsqrt_kernel(const float* __restrict__ deg,
                             float* __restrict__ dinv, int n) {
  int i = blockIdx.x * blockDim.x + threadIdx.x;
  if (i < n) dinv[i] = rsqrtf(deg[i]);
}

// ---------------- WMMA GEMM: h = x @ W  (f16 in, f32 acc) ----------------
// Block = 256 threads = 8 waves; each wave computes a 16x128 slice of h.
// A fragment (16x32 f16): lane = M (lanes 0-15 == 16-31 in M), K half split
//   by lane>=16; elements 0..7 = K in [koff, koff+8), 8..15 = K in [16+koff,...).
// B fragment (32x16 f16): lane&15 = N column, K split by lane>=16 with the same element order as A.
// D (16x16 f32): VGPR r -> M = r + 8*(lane>=16), N = lane&15.

__global__ __launch_bounds__(256)
void gemm_wmma_f16_kernel(const float* __restrict__ x,
                          const float* __restrict__ w,
                          float* __restrict__ h, int n_nodes) {
  __shared__ _Float16 wlds[128 * LDS_PITCH];

  const int tid = threadIdx.x;

  // Stage W (128x128 f32 -> f16) into padded LDS.
  for (int idx = tid; idx < NFEAT * NFEAT; idx += 256) {
    int r = idx >> 7;
    int c = idx & 127;
    wlds[r * LDS_PITCH + c] = (_Float16)w[idx];
  }

  const int wave = tid >> 5;
  const int lane = tid & 31;
  const int mrow = lane & 15;
  const int koff = (lane >> 4) << 3;  // 0 for lanes 0-15, 8 for lanes 16-31
  const int mhi  = (lane >> 4) << 3;  // D-matrix M offset for upper half-wave

  const int row_base = blockIdx.x * 128 + wave * 16;            // fits int32
  int row = row_base + mrow;
  int rclamp = (row < n_nodes) ? row : (n_nodes - 1);
  const float* xr = x + rclamp * NFEAT;
  __builtin_prefetch((const void*)xr, 0, 0);  // global_prefetch_b8

  const bool full_tile = (row_base + 16) <= n_nodes;

  __syncthreads();

  // Build the 4 A fragments (K = 0..127 in 4 tiles of 32), kept in VGPRs.
  v16h a[4];
  #pragma unroll
  for (int kt = 0; kt < 4; ++kt) {
    const int k0 = kt * 32 + koff;
    v4f f0 = *(const v4f*)(xr + k0);
    v4f f1 = *(const v4f*)(xr + k0 + 4);
    v4f f2 = *(const v4f*)(xr + k0 + 16);
    v4f f3 = *(const v4f*)(xr + k0 + 20);
    #pragma unroll
    for (int j = 0; j < 4; ++j) {
      a[kt][j]      = (_Float16)f0[j];
      a[kt][4 + j]  = (_Float16)f1[j];
      a[kt][8 + j]  = (_Float16)f2[j];
      a[kt][12 + j] = (_Float16)f3[j];
    }
  }

  // 8 N-tiles of 16 columns each.
  #pragma unroll
  for (int nt = 0; nt < 8; ++nt) {
    v8f acc = {};
    #pragma unroll
    for (int kt = 0; kt < 4; ++kt) {
      const _Float16* bsrc = &wlds[(kt * 32 + koff) * LDS_PITCH + nt * 16 + (lane & 15)];
      union { v16h v; v8h half8[2]; } bu;
      #pragma unroll
      for (int s = 0; s < 8; ++s) {
        bu.v[s]     = bsrc[s * LDS_PITCH];
        bu.v[8 + s] = bsrc[(16 + s) * LDS_PITCH];
      }
      acc = __builtin_amdgcn_wmma_f32_16x16x32_f16(
          /*neg_a=*/false, a[kt], /*neg_b=*/false, bu.v,
          /*c_mod=*/(short)0, acc, /*reuse_a=*/false, /*reuse_b=*/false);
    }
    // Store D: VGPR r -> row (row_base + mhi + r), col nt*16 + (lane&15).
    const int ncol = nt * 16 + (lane & 15);
    float* hp = h + (row_base + mhi) * NFEAT + ncol;
    if (full_tile) {
      #pragma unroll
      for (int r = 0; r < 8; ++r) hp[r * NFEAT] = acc[r];
    } else {
      #pragma unroll
      for (int r = 0; r < 8; ++r) {
        if (row_base + mhi + r < n_nodes) hp[r * NFEAT] = acc[r];
      }
    }
  }
}

// ---------------- edge scatter: one wave32 per edge ----------------
// Each lane moves 4 floats (32 lanes x 4 = 128 features per edge).

__global__ __launch_bounds__(256)
void scatter_kernel(const long long* __restrict__ row,
                    const long long* __restrict__ col,
                    const float* __restrict__ dinv,
                    const float* __restrict__ h,
                    float* __restrict__ out, int e) {
  const int w = (blockIdx.x * 256 + threadIdx.x) >> 5;
  const int lane = threadIdx.x & 31;
  if (w >= e) return;
  const int r = (int)row[w];
  const int c = (int)col[w];
  const float nrm = dinv[r] * dinv[c];
  const v4f hv = *(const v4f*)(h + r * NFEAT + lane * 4);
  float* ob = out + c * NFEAT + lane * 4;
  atomicAdd(ob + 0, hv.x * nrm);
  atomicAdd(ob + 1, hv.y * nrm);
  atomicAdd(ob + 2, hv.z * nrm);
  atomicAdd(ob + 3, hv.w * nrm);
}

// ---------------- finalize: self-loop + bias + ReLU ----------------

__global__ void finalize_kernel(const float* __restrict__ h,
                                const float* __restrict__ dinv,
                                const float* __restrict__ bias,
                                float* __restrict__ out, int tot) {
  int i = blockIdx.x * blockDim.x + threadIdx.x;
  if (i >= tot) return;
  int node = i >> 7;
  int f = i & 127;
  float d = dinv[node];
  float v = out[i] + h[i] * d * d + bias[f];
  out[i] = v > 0.0f ? v : 0.0f;
}

// ---------------- launcher ----------------

extern "C" void kernel_launch(void* const* d_in, const int* in_sizes, int n_in,
                              void* d_out, int out_size, void* d_ws, size_t ws_size,
                              hipStream_t stream) {
  const float*     x    = (const float*)d_in[0];
  const long long* ei   = (const long long*)d_in[1];
  const float*     w    = (const float*)d_in[2];
  const float*     bias = (const float*)d_in[3];
  float*           out  = (float*)d_out;

  const int n = in_sizes[0] / NFEAT;   // 50000
  const int e = in_sizes[1] / 2;       // 800000
  const long long* rowp = ei;          // source nodes
  const long long* colp = ei + e;      // target nodes

  // Workspace layout: h [n*128] | deg [n] | dinv [n]
  float* h    = (float*)d_ws;
  float* deg  = h + (size_t)n * NFEAT;
  float* dinv = deg + n;

  const int tot = n * NFEAT;

  zero_f32_kernel<<<(tot + 255) / 256, 256, 0, stream>>>(out, tot);
  init_deg_kernel<<<(n + 255) / 256, 256, 0, stream>>>(deg, n);
  count_deg_kernel<<<(e + 255) / 256, 256, 0, stream>>>(colp, deg, e);
  rsqrt_kernel<<<(n + 255) / 256, 256, 0, stream>>>(deg, dinv, n);

  gemm_wmma_f16_kernel<<<(n + 127) / 128, 256, 0, stream>>>(x, w, h, n);

  scatter_kernel<<<(e + 7) / 8, 256, 0, stream>>>(rowp, colp, dinv, h, out, e);

  finalize_kernel<<<(tot + 255) / 256, 256, 0, stream>>>(h, dinv, bias, out, tot);
}
